// MHA_GCN_28389733827055
// MI455X (gfx1250) — matrix operand order repaired and checked by reading, not compiled, not run on hardware
//
#include <hip/hip_runtime.h>
#include <hip/hip_bf16.h>

// ---------------------------------------------------------------------------
// Problem constants (from the reference)
// ---------------------------------------------------------------------------
#define NN      4096
#define F_IN    128
#define G1      128
#define G2      512
#define E3      1536      // 3*G2
#define HEADS   4
#define HD      128       // head dim

typedef __attribute__((ext_vector_type(16))) _Float16 v16h;
typedef __attribute__((ext_vector_type(8)))  _Float16 h8;
typedef __attribute__((ext_vector_type(8)))  float    v8f;

// ---------------------------------------------------------------------------
// WMMA helpers (gfx1250, wave32)
// ---------------------------------------------------------------------------
__device__ __forceinline__ v8f wmma32(v16h a, v16h b, v8f c) {
    return __builtin_amdgcn_wmma_f32_16x16x32_f16(
        /*neg_a=*/false, a, /*neg_b=*/false, b,
        /*c_mod=*/(short)0, c, /*reuse_a=*/false, /*reuse_b=*/false);
}

// A-fragment (16x32 f16, row-major source): lane m = lane&15,
// element e -> K = k0 + (lane>=16?16:0) + e  (16 contiguous halves)
// caller passes p = &A[m*ldk + k0 + hi*16]
__device__ __forceinline__ v16h frag_a(const _Float16* p) {
    h8 lo = *(const h8*)(p);
    h8 hi = *(const h8*)(p + 8);
    return __builtin_shufflevector(lo, hi, 0,1,2,3,4,5,6,7,8,9,10,11,12,13,14,15);
}

// B-fragment (32x16 f16) when B is stored transposed [N x K]: lane n = lane&15,
// element e -> K = k0 + (lane>=16?16:0) + e  (16 contiguous halves)
// caller passes p = &Bt[n*K + k0 + hi*16]
__device__ __forceinline__ v16h frag_bt(const _Float16* p) {
    h8 lo = *(const h8*)(p);
    h8 hi = *(const h8*)(p + 8);
    return __builtin_shufflevector(lo, hi, 0,1,2,3,4,5,6,7,8,9,10,11,12,13,14,15);
}

__device__ __forceinline__ float lane_bcast(float x, int srcLane) {
    int v = __builtin_amdgcn_ds_bpermute(srcLane << 2, __builtin_bit_cast(int, x));
    return __builtin_bit_cast(float, v);
}

// ---------------------------------------------------------------------------
// 1) degree -> D^{-1/2}   (one 256-thread block per row)
// ---------------------------------------------------------------------------
__global__ __launch_bounds__(256)
void degree_kernel(const float* __restrict__ adj, float* __restrict__ dinv) {
    __shared__ float red[256];
    const int row = blockIdx.x;
    float s = 0.f;
    for (int j = threadIdx.x; j < NN; j += 256) s += adj[(size_t)row * NN + j];
    red[threadIdx.x] = s;
    __syncthreads();
    for (int st = 128; st > 0; st >>= 1) {
        if (threadIdx.x < st) red[threadIdx.x] += red[threadIdx.x + st];
        __syncthreads();
    }
    if (threadIdx.x == 0) {
        float deg = red[0] + 1.0f;                 // self loop
        dinv[row] = (deg > 0.f) ? rsqrtf(deg) : 0.f;
    }
}

// ---------------------------------------------------------------------------
// 2) A_h = fp16( dinv[i] * (adj + I) * dinv[j] )
// ---------------------------------------------------------------------------
__global__ __launch_bounds__(256)
void normadj_kernel(const float* __restrict__ adj, const float* __restrict__ dinv,
                    _Float16* __restrict__ Ah) {
    size_t total = (size_t)NN * NN;
    for (size_t idx = (size_t)blockIdx.x * blockDim.x + threadIdx.x;
         idx < total; idx += (size_t)gridDim.x * blockDim.x) {
        int i = (int)(idx >> 12);
        int j = (int)(idx & (NN - 1));
        float v = adj[idx] + ((i == j) ? 1.0f : 0.0f);
        Ah[idx] = (_Float16)(dinv[i] * dinv[j] * v);
    }
}

// ---------------------------------------------------------------------------
// 3) transpose-convert f32 [R x C] -> f16 [C x R]
// ---------------------------------------------------------------------------
__global__ __launch_bounds__(256)
void tconv_kernel(const float* __restrict__ src, _Float16* __restrict__ dst,
                  int R, int C) {
    size_t total = (size_t)R * C;
    for (size_t idx = (size_t)blockIdx.x * blockDim.x + threadIdx.x;
         idx < total; idx += (size_t)gridDim.x * blockDim.x) {
        int r = (int)(idx / C);
        int c = (int)(idx - (size_t)r * C);
        dst[(size_t)c * R + r] = (_Float16)src[idx];
    }
}

// ---------------------------------------------------------------------------
// 4) WMMA GEMM:  C[MxN] = A_h[MxK] @ Bt_h[NxK]^T (+bias) (relu?)
//    outputs: Cf (f32, optional), Ch (f16, optional, row-major or transposed)
//    block = 128 threads (4 waves); wave tile = 32x32 (2x2 WMMA tiles)
//    requires M %128==0, N%32==0, K%32==0
// ---------------------------------------------------------------------------
__global__ __launch_bounds__(128)
void gemm_h_kernel(const _Float16* __restrict__ A, const _Float16* __restrict__ Bt,
                   const float* __restrict__ bias,
                   float* __restrict__ Cf, _Float16* __restrict__ Ch,
                   int M, int N, int K, int relu, int transCh) {
    const int lane = threadIdx.x & 31;
    const int wave = threadIdx.x >> 5;
    const int lo16 = lane & 15;
    const int hi   = lane >> 4;               // 0 or 1
    const int m0 = blockIdx.y * 128 + wave * 32;
    const int n0 = blockIdx.x * 32;

    v8f acc00 = {}, acc01 = {}, acc10 = {}, acc11 = {};

    const _Float16* Ar0 = A  + (size_t)(m0 + lo16) * K + hi * 16;
    const _Float16* Ar1 = Ar0 + (size_t)16 * K;
    const _Float16* Br0 = Bt + (size_t)(n0 + lo16) * K + hi * 16;
    const _Float16* Br1 = Br0 + (size_t)16 * K;

    for (int k0 = 0; k0 < K; k0 += 32) {
        __builtin_prefetch(Ar0 + k0 + 256, 0, 0);   // global_prefetch_b8
        __builtin_prefetch(Br0 + k0 + 256, 0, 0);
        v16h a0 = frag_a(Ar0 + k0);
        v16h a1 = frag_a(Ar1 + k0);
        v16h b0 = frag_bt(Br0 + k0);
        v16h b1 = frag_bt(Br1 + k0);
        acc00 = wmma32(a0, b0, acc00);
        acc01 = wmma32(a0, b1, acc01);
        acc10 = wmma32(a1, b0, acc10);
        acc11 = wmma32(a1, b1, acc11);
    }

    // epilogue: lane holds col (n-tile base + lo16), rows r + 8*hi
    v8f* accs[4] = { &acc00, &acc01, &acc10, &acc11 };
#pragma unroll
    for (int mt = 0; mt < 2; ++mt) {
#pragma unroll
        for (int nt = 0; nt < 2; ++nt) {
            v8f& a = *accs[mt * 2 + nt];
            int col = n0 + nt * 16 + lo16;
            float bv = bias ? bias[col] : 0.0f;
#pragma unroll
            for (int r = 0; r < 8; ++r) {
                int row = m0 + mt * 16 + r + 8 * hi;
                float v = a[r] + bv;
                if (relu) v = fmaxf(v, 0.0f);
                if (Cf) Cf[(size_t)row * N + col] = v;
                if (Ch) {
                    if (transCh) Ch[(size_t)col * M + row] = (_Float16)v;
                    else         Ch[(size_t)row * N + col] = (_Float16)v;
                }
            }
        }
    }
}

// ---------------------------------------------------------------------------
// 5) Flash attention, one wave per (16-query block, head).
//    QKV layout: [N x 1536] f16, Q at +0, K at +512, V at +1024, head h at h*128.
//    Computes S^T = K_blk @ Q^T so each lane owns one query's scores.
//    ctx written as f16 [N x 512].
// ---------------------------------------------------------------------------
__global__ __launch_bounds__(32)
void attn_kernel(const _Float16* __restrict__ QKV, _Float16* __restrict__ ctx) {
    const int lane = threadIdx.x & 31;
    const int lo16 = lane & 15;
    const int hi   = lane >> 4;
    const int h    = blockIdx.y;
    const int qb   = blockIdx.x * 16;
    const float scale = 0.088388347648318447f;   // 1/sqrt(128)

    // Q fragments (B operand of S^T), all 4 K-chunks kept resident
    const _Float16* Qrow = QKV + (size_t)(qb + lo16) * E3 + h * HD;
    v16h qf[4];
#pragma unroll
    for (int kc = 0; kc < 4; ++kc) {
        const _Float16* p = Qrow + kc * 32 + hi * 16;
        h8 l0 = *(const h8*)(p);
        h8 l1 = *(const h8*)(p + 8);
        qf[kc] = __builtin_shufflevector(l0, l1, 0,1,2,3,4,5,6,7,8,9,10,11,12,13,14,15);
    }

    float m_run = -1e30f, l_run = 0.0f;
    v8f o[8];
#pragma unroll
    for (int d = 0; d < 8; ++d) o[d] = (v8f){};

    for (int kb = 0; kb < NN; kb += 32) {
        // ---- S^T for two 16-key tiles -------------------------------------
        v8f st[2];
#pragma unroll
        for (int t = 0; t < 2; ++t) {
            v8f s = {};
            const _Float16* Krow =
                QKV + (size_t)(kb + t * 16 + lo16) * E3 + G2 + h * HD + hi * 16;
#pragma unroll
            for (int kc = 0; kc < 4; ++kc)
                s = wmma32(frag_a(Krow + kc * 32), qf[kc], s);
            st[t] = s;
        }
        // ---- online softmax (lane owns query lo16; other 8 keys in lane^16)
        float cmax = -1e30f;
#pragma unroll
        for (int r = 0; r < 8; ++r) {
            cmax = fmaxf(cmax, st[0][r] * scale);
            cmax = fmaxf(cmax, st[1][r] * scale);
        }
        cmax = fmaxf(cmax, lane_bcast(cmax, lane ^ 16));
        float newm = fmaxf(m_run, cmax);
        float corr = __expf(m_run - newm);

        v16h pa;                 // P fragment: maps 1:1 onto A layout for P@V
        float psum = 0.0f;
#pragma unroll
        for (int r = 0; r < 8; ++r) {
            float p0 = __expf(st[0][r] * scale - newm);
            float p1 = __expf(st[1][r] * scale - newm);
            psum += p0 + p1;
            pa[r]     = (_Float16)p0;
            pa[8 + r] = (_Float16)p1;
        }
        psum += lane_bcast(psum, lane ^ 16);
        l_run = l_run * corr + psum;
        m_run = newm;

        // rescale ctx accumulators by the per-row correction
        float crow[8];
#pragma unroll
        for (int r = 0; r < 8; ++r) crow[r] = lane_bcast(corr, r + 8 * hi);
#pragma unroll
        for (int d = 0; d < 8; ++d)
#pragma unroll
            for (int r = 0; r < 8; ++r) o[d][r] *= crow[r];

        // ---- ctx += P(16x32) @ V(32x16) for each of 8 head-dim chunks -----
#pragma unroll
        for (int d = 0; d < 8; ++d) {
            const _Float16* Vp =
                QKV + (size_t)(kb + hi * 8) * E3 + 2 * G2 + h * HD + d * 16 + lo16;
            v16h bv;
#pragma unroll
            for (int e = 0; e < 16; ++e) bv[e] = Vp[(size_t)(e + (e >> 3) * 8) * E3];
            o[d] = wmma32(pa, bv, o[d]);
        }
    }

    // ---- finalize: divide by l, store f16 ctx -----------------------------
    float lrow[8];
#pragma unroll
    for (int r = 0; r < 8; ++r) lrow[r] = lane_bcast(l_run, r + 8 * hi);
#pragma unroll
    for (int d = 0; d < 8; ++d) {
#pragma unroll
        for (int r = 0; r < 8; ++r) {
            int row = qb + r + 8 * hi;
            int col = h * HD + d * 16 + lo16;
            ctx[(size_t)row * G2 + col] = (_Float16)(o[d][r] / lrow[r]);
        }
    }
}

// ---------------------------------------------------------------------------
// 6) column mean over 4096 rows  (g[512])
// ---------------------------------------------------------------------------
__global__ __launch_bounds__(256)
void colmean_kernel(const float* __restrict__ X, float* __restrict__ g) {
    int col = blockIdx.x * blockDim.x + threadIdx.x;
    if (col < G2) {
        float s = 0.f;
        for (int r = 0; r < NN; ++r) s += X[(size_t)r * G2 + col];
        g[col] = s * (1.0f / NN);
    }
}

// ---------------------------------------------------------------------------
// 7) final FC:  out[2] = g @ fc_w + fc_b
// ---------------------------------------------------------------------------
__global__ __launch_bounds__(512)
void final_kernel(const float* __restrict__ g, const float* __restrict__ fcw,
                  const float* __restrict__ fcb, float* __restrict__ out) {
    __shared__ float r0[512], r1[512];
    int d = threadIdx.x;
    float gv = g[d];
    r0[d] = gv * fcw[d * 2 + 0];
    r1[d] = gv * fcw[d * 2 + 1];
    __syncthreads();
    for (int s = 256; s > 0; s >>= 1) {
        if (d < s) { r0[d] += r0[d + s]; r1[d] += r1[d + s]; }
        __syncthreads();
    }
    if (d == 0) { out[0] = r0[0] + fcb[0]; out[1] = r1[0] + fcb[1]; }
}

// ---------------------------------------------------------------------------
// Host orchestration
// ---------------------------------------------------------------------------
extern "C" void kernel_launch(void* const* d_in, const int* in_sizes, int n_in,
                              void* d_out, int out_size, void* d_ws, size_t ws_size,
                              hipStream_t stream) {
    (void)in_sizes; (void)n_in; (void)out_size; (void)ws_size;
    const float* x    = (const float*)d_in[0];
    const float* adj  = (const float*)d_in[1];
    const float* W1   = (const float*)d_in[2];
    const float* b1   = (const float*)d_in[3];
    const float* W2   = (const float*)d_in[4];
    const float* b2   = (const float*)d_in[5];
    const float* inW  = (const float*)d_in[6];
    const float* inB  = (const float*)d_in[7];
    const float* outW = (const float*)d_in[8];
    const float* outB = (const float*)d_in[9];
    const float* fcw  = (const float*)d_in[10];
    const float* fcb  = (const float*)d_in[11];
    float* out = (float*)d_out;

    char* wsp = (char*)d_ws;
    size_t off = 0;
    auto carve = [&](size_t bytes) -> char* {
        char* p = wsp + off;
        off += (bytes + 255) & ~(size_t)255;
        return p;
    };
    _Float16* Ah     = (_Float16*)carve((size_t)NN * NN * 2);      // 32 MB
    _Float16* Xt     = (_Float16*)carve((size_t)F_IN * NN * 2);
    _Float16* W1t    = (_Float16*)carve((size_t)G1 * F_IN * 2);
    _Float16* W2t    = (_Float16*)carve((size_t)G2 * G1 * 2);
    _Float16* inWt   = (_Float16*)carve((size_t)E3 * G2 * 2);
    _Float16* outWt  = (_Float16*)carve((size_t)G2 * G2 * 2);
    _Float16* S1h    = (_Float16*)carve((size_t)NN * G1 * 2);
    _Float16* H1t    = (_Float16*)carve((size_t)G1 * NN * 2);
    _Float16* S2h    = (_Float16*)carve((size_t)NN * G1 * 2);
    _Float16* H2h    = (_Float16*)carve((size_t)NN * G2 * 2);
    _Float16* QKVh   = (_Float16*)carve((size_t)NN * E3 * 2);      // 12 MB
    _Float16* CTXh   = (_Float16*)carve((size_t)NN * G2 * 2);
    float*    OUTf   = (float*)   carve((size_t)NN * G2 * 4);      // 8 MB
    float*    dinv   = (float*)   carve((size_t)NN * 4);
    float*    gmean  = (float*)   carve((size_t)G2 * 4);

    // --- normalization data -------------------------------------------------
    degree_kernel<<<NN, 256, 0, stream>>>(adj, dinv);
    normadj_kernel<<<16384, 256, 0, stream>>>(adj, dinv, Ah);

    // --- f16 transposed operands -------------------------------------------
    tconv_kernel<<<2048, 256, 0, stream>>>(x,    Xt,   NN,  F_IN);
    tconv_kernel<<<64,   256, 0, stream>>>(W1,   W1t,  F_IN, G1);
    tconv_kernel<<<256,  256, 0, stream>>>(W2,   W2t,  G1,  G2);
    tconv_kernel<<<3072, 256, 0, stream>>>(inW,  inWt, G2,  E3);
    tconv_kernel<<<1024, 256, 0, stream>>>(outW, outWt,G2,  G2);

    // --- GCN layer 1: S1 = Ah @ X ; H1 = relu(S1 @ W1 + b1) (stored [G1 x N])
    gemm_h_kernel<<<dim3(F_IN/32, NN/128), 128, 0, stream>>>(
        Ah, Xt, nullptr, nullptr, S1h, NN, F_IN, NN, 0, 0);
    gemm_h_kernel<<<dim3(G1/32, NN/128), 128, 0, stream>>>(
        S1h, W1t, b1, nullptr, H1t, NN, G1, F_IN, 1, 1);

    // --- GCN layer 2: S2 = Ah @ H1 ; H2 = S2 @ W2 + b2 ---------------------
    gemm_h_kernel<<<dim3(G1/32, NN/128), 128, 0, stream>>>(
        Ah, H1t, nullptr, nullptr, S2h, NN, G1, NN, 0, 0);
    gemm_h_kernel<<<dim3(G2/32, NN/128), 128, 0, stream>>>(
        S2h, W2t, b2, nullptr, H2h, NN, G2, G1, 0, 0);

    // --- MHA: QKV projection, flash attention, out projection --------------
    gemm_h_kernel<<<dim3(E3/32, NN/128), 128, 0, stream>>>(
        H2h, inWt, inB, nullptr, QKVh, NN, E3, G2, 0, 0);
    attn_kernel<<<dim3(NN/16, HEADS), 32, 0, stream>>>(QKVh, CTXh);
    gemm_h_kernel<<<dim3(G2/32, NN/128), 128, 0, stream>>>(
        CTXh, outWt, outB, OUTf, nullptr, NN, G2, G2, 0, 0);

    // --- mean pool + classifier --------------------------------------------
    colmean_kernel<<<2, 256, 0, stream>>>(OUTf, gmean);
    final_kernel<<<1, 512, 0, stream>>>(gmean, fcw, fcb, out);
}
